// DecomposeNet_51737176048111
// MI455X (gfx1250) — matrix operand order repaired and checked by reading, not compiled, not run on hardware
//
#include <hip/hip_runtime.h>

// ---------------------------------------------------------------------------
// CDNA5 (gfx1250) wave32 WMMA types
// ---------------------------------------------------------------------------
typedef __attribute__((ext_vector_type(16))) _Float16 v16h;
typedef __attribute__((ext_vector_type(8)))  float    v8f;
typedef __attribute__((ext_vector_type(4)))  int      v4i;

#define CF_RELU_IN 1
#define CF_ADD_RES 2
#define CF_TRANS   4

// Async global->LDS copies (CDNA5: GLOBAL_LOAD_ASYNC_TO_LDS_B128, ASYNCcnt)
#if defined(__has_builtin)
#if __has_builtin(__builtin_amdgcn_global_load_async_to_lds_b128) && \
    __has_builtin(__builtin_amdgcn_s_wait_asynccnt)
#define HAVE_ASYNC_LDS 1
#endif
#endif
#ifndef HAVE_ASYNC_LDS
#define HAVE_ASYNC_LDS 0
#endif

#if HAVE_ASYNC_LDS
// Builtin signature (from hipcc diagnostic): param0 = v4i AS1* (global src),
// param1 = v4i AS3* (LDS dst), then two immediate ints (offset, cpol).
// Cast via integer: AS1 shares the 64-bit VA; AS3 = addr[31:0] per ISA.
__device__ __forceinline__ void async_copy_b128(const float* g, float* l) {
    __builtin_amdgcn_global_load_async_to_lds_b128(
        (__attribute__((address_space(1))) v4i*)(unsigned long long)g,
        (__attribute__((address_space(3))) v4i*)(unsigned long long)l,
        0, 0);
}
#endif

// ---------------------------------------------------------------------------
// Implicit-GEMM conv / conv-transpose via V_WMMA_F32_16X16X32_F16 (wave32).
// GEMM: M = N*Hout*Wout, Ndim = Cout, K = Cin*KH*KW.
// Block = 128 threads = 4 waves; block covers 16 M-rows x 128 Cout columns.
// Each wave owns two 16x16 D tiles (2 accumulators -> 2 WMMAs per K-chunk).
// A tile (16x32 f16) and B tile (128x32 f32, rows padded to 36 floats for
// 16B-aligned async segments + conflict-free reads) are staged in LDS.
// Fragment lane layouts per ISA 7.12.2 (wave32):
//   A: lane l, elem e -> m=l&15, k=(l>>4)*8 + (e&7) + (e>>3)*16
//   B: lane l, elem e -> n=l&15, k=(l>>4)*8 + (e&7) + (e>>3)*16
//   C/D: lane l, vgpr r -> m=(l>>4)*8+r, n=l&15
// ---------------------------------------------------------------------------
__global__ void __launch_bounds__(128)
conv2d_wmma_kernel(const float* __restrict__ in, const float* __restrict__ wgt,
                   const float* __restrict__ bias, const float* __restrict__ res,
                   float* __restrict__ out,
                   int Nb, int Cin, int Hin, int Win,
                   int Cout, int Hout, int Wout,
                   int KH, int KW, int stride, int pad, int flags)
{
    const int lane = threadIdx.x & 31;
    const int wave = threadIdx.x >> 5;
    const int Mtot = Nb * Hout * Wout;
    const int khkw = KH * KW;
    const int Ktot = Cin * khkw;
    const int m0   = blockIdx.x * 16;
    const int bn0  = blockIdx.y * 128;          // block Cout base
    const int n0   = bn0 + wave * 32;           // wave's first 16-wide tile
    const int HWo  = Hout * Wout;
    const int HWi  = Hin * Win;
    const bool trans  = (flags & CF_TRANS) != 0;
    const bool reluin = (flags & CF_RELU_IN) != 0;

    __shared__ _Float16 Atile[16][40];
    __shared__ __align__(16) float Btile[128][36];   // 18,432 B

    // ---- hoist K-invariant M-row decode for the A-tile gather ----
    int arow[4], akk[4], ab_[4], aoh[4], aow[4]; bool amv[4];
    #pragma unroll
    for (int i = 0; i < 4; ++i) {
        int e = threadIdx.x + i * 128;          // 0..511
        arow[i] = e >> 5; akk[i] = e & 31;
        int m = m0 + arow[i];
        amv[i] = m < Mtot;
        int mm = amv[i] ? m : 0;
        int b  = mm / HWo; int hw = mm - b * HWo;
        ab_[i] = b; aoh[i] = hw / Wout; aow[i] = hw - aoh[i] * Wout;
    }

    v8f acc0 = {}, acc1 = {};
    for (int kc = 0; kc < Ktot; kc += 32) {
        // ---- stage B tile: 128 n x 32 k (f32) ----
        // 1024 16B segments; 8 per thread. Non-transposed weights are
        // contiguous in k (w[n*Ktot + k]) -> async B128 copies.
        #pragma unroll
        for (int j = 0; j < 8; ++j) {
            int s    = threadIdx.x + j * 128;   // 0..1023
            int nl   = s >> 3, part = s & 7;
            int n    = bn0 + nl;
            int kb   = kc + part * 4;
            bool fast = !trans && n < Cout && (kb + 4) <= Ktot;
            if (fast) {
#if HAVE_ASYNC_LDS
                async_copy_b128(&wgt[(size_t)n * Ktot + kb], &Btile[nl][part * 4]);
#else
                const float* g = &wgt[(size_t)n * Ktot + kb];
                #pragma unroll
                for (int t = 0; t < 4; ++t) Btile[nl][part * 4 + t] = g[t];
#endif
            } else {
                #pragma unroll
                for (int t = 0; t < 4; ++t) {
                    int k = kb + t;
                    float v = 0.0f;
                    if (n < Cout && k < Ktot) {
                        if (trans) {
                            int ci = k / khkw; int r = k - ci * khkw;
                            v = wgt[((size_t)ci * Cout + n) * khkw + r]; // (Cin,Cout,kh,kw)
                        } else {
                            v = wgt[(size_t)n * Ktot + k];
                        }
                    }
                    Btile[nl][part * 4 + t] = v;
                }
            }
        }
        // ---- stage A tile (gather with boundary/ReLU) ----
        #pragma unroll
        for (int i = 0; i < 4; ++i) {
            int k = kc + akk[i];
            float v = 0.0f;
            if (amv[i] && k < Ktot) {
                int ci, ky, kx;
                if (khkw == 1) { ci = k; ky = 0; kx = 0; }
                else { ci = k / khkw; int r = k - ci * khkw; ky = r / KW; kx = r - ky * KW; }
                int iy, ix; bool ok;
                if (!trans) {
                    iy = aoh[i] * stride - pad + ky;
                    ix = aow[i] * stride - pad + kx;
                    ok = (iy >= 0) & (iy < Hin) & (ix >= 0) & (ix < Win);
                } else {
                    int ny = aoh[i] + pad - ky, nx = aow[i] + pad - kx;
                    ok = (ny >= 0) & (nx >= 0) &&
                         (ny % stride) == 0 && (nx % stride) == 0;
                    iy = ny / stride; ix = nx / stride;
                    ok = ok && (iy < Hin) && (ix < Win);
                }
                if (ok) {
                    v = in[((size_t)(ab_[i] * Cin + ci)) * HWi + (size_t)iy * Win + ix];
                    if (reluin) v = v > 0.0f ? v : 0.0f;
                }
            }
            Atile[arow[i]][akk[i]] = (_Float16)v;
        }
#if HAVE_ASYNC_LDS
        __builtin_amdgcn_s_wait_asynccnt(0);
#endif
        __syncthreads();
        // ---- fragments ----
        v16h a;
        {
            int mrow = lane & 15;
            int kb   = (lane >> 4) * 8;
            #pragma unroll
            for (int e = 0; e < 16; ++e)
                a[e] = Atile[mrow][kb + (e & 7) + ((e >> 3) << 4)];
        }
        v16h b0, b1;
        {
            int nl = wave * 32 + (lane & 15);
            int kb = (lane >> 4) * 8;
            #pragma unroll
            for (int e = 0; e < 16; ++e) {
                b0[e] = (_Float16)Btile[nl][kb + (e & 7) + ((e >> 3) << 4)];
                b1[e] = (_Float16)Btile[nl + 16][kb + (e & 7) + ((e >> 3) << 4)];
            }
        }
        acc0 = __builtin_amdgcn_wmma_f32_16x16x32_f16(false, a, false, b0, (short)0, acc0, false, false);
        acc1 = __builtin_amdgcn_wmma_f32_16x16x32_f16(false, a, false, b1, (short)0, acc1, false, false);
        __syncthreads();
    }
    // ---- epilogue: incremental (b,hw) decode, bias (+residual) ----
    int nA = n0 + (lane & 15);
    int nB = nA + 16;
    float bvA = (nA < Cout) ? bias[nA] : 0.0f;
    float bvB = (nB < Cout) ? bias[nB] : 0.0f;
    int mbase = m0 + (lane >> 4) * 8;
    int bb = mbase / HWo;
    int hw = mbase - bb * HWo;
    #pragma unroll
    for (int r = 0; r < 8; ++r) {
        if (hw >= HWo) { hw -= HWo; ++bb; }
        if (mbase + r < Mtot) {
            if (nA < Cout) {
                size_t o = ((size_t)(bb * Cout + nA)) * HWo + hw;
                float v = acc0[r] + bvA;
                if (flags & CF_ADD_RES) v += res[o];
                out[o] = v;
            }
            if (nB < Cout) {
                size_t o = ((size_t)(bb * Cout + nB)) * HWo + hw;
                float v = acc1[r] + bvB;
                if (flags & CF_ADD_RES) v += res[o];
                out[o] = v;
            }
        }
        ++hw;
    }
}

// ---------------------------------------------------------------------------
// VQ nearest-embed argmin: per wave, 16 positions x 512 codes.
// score = ||w||^2 - 2 x.w  (||x||^2 constant per row).  x.w via WMMA.
// ---------------------------------------------------------------------------
__global__ void __launch_bounds__(32)
vq_argmin_kernel(const float* __restrict__ x, const float* __restrict__ cb,
                 const float* __restrict__ wnorm,
                 int Bn, int C, int HW, int Kc, int* __restrict__ idx_out)
{
    const int lane = threadIdx.x & 31;
    const int m0   = blockIdx.x * 16;
    const int M    = Bn * HW;
    const int nch  = C >> 5;                 // 4 (C=128) or 8 (C=256)
    v16h afrag[8];

    int mrow = m0 + (lane & 15);
    bool mok = mrow < M;
    int bb = 0, hw = 0;
    if (mok) { bb = mrow / HW; hw = mrow - bb * HW; }
    int kb = (lane >> 4) * 8;
    #pragma unroll
    for (int c = 0; c < 8; ++c) {
        if (c < nch) {
            #pragma unroll
            for (int e = 0; e < 16; ++e) {
                int k = c * 32 + kb + (e & 7) + ((e >> 3) << 4);
                float v = 0.0f;
                if (mok) v = x[((size_t)(bb * C + k)) * HW + hw];
                afrag[c][e] = (_Float16)v;
            }
        }
    }

    float bestd[8]; int bestc[8];
    #pragma unroll
    for (int r = 0; r < 8; ++r) { bestd[r] = 3.4e38f; bestc[r] = 0; }

    for (int ct = 0; ct < Kc / 16; ++ct) {
        v8f acc = {};
        int n = ct * 16 + (lane & 15);
        #pragma unroll
        for (int c = 0; c < 8; ++c) {
            if (c < nch) {
                v16h bf;
                int kbase = c * 32 + (lane >> 4) * 8;
                #pragma unroll
                for (int e = 0; e < 16; ++e)
                    bf[e] = (_Float16)cb[(size_t)(kbase + (e & 7) + ((e >> 3) << 4)) * Kc + n];
                acc = __builtin_amdgcn_wmma_f32_16x16x32_f16(
                          false, afrag[c], false, bf, (short)0, acc, false, false);
            }
        }
        float wn = wnorm[n];
        #pragma unroll
        for (int r = 0; r < 8; ++r) {
            float d = wn - 2.0f * acc[r];
            if (d < bestd[r]) { bestd[r] = d; bestc[r] = n; }
        }
    }
    #pragma unroll
    for (int off = 1; off < 16; off <<= 1) {
        #pragma unroll
        for (int r = 0; r < 8; ++r) {
            float od = __shfl_xor(bestd[r], off, 32);
            int   oc = __shfl_xor(bestc[r], off, 32);
            if (od < bestd[r] || (od == bestd[r] && oc < bestc[r])) {
                bestd[r] = od; bestc[r] = oc;
            }
        }
    }
    if ((lane & 15) == 0) {
        #pragma unroll
        for (int r = 0; r < 8; ++r) {
            int m = m0 + (lane >> 4) * 8 + r;
            if (m < M) idx_out[m] = bestc[r];
        }
    }
}

// q[b,c,h,w] = codebook[c, idx[b,h,w]]  (2D grid: no per-lane div/mod)
__global__ void vq_gather_kernel(const float* __restrict__ cb, const int* __restrict__ idx,
                                 float* __restrict__ q, int C, int HW, int Kc)
{
    int hw = blockIdx.x * blockDim.x + threadIdx.x;
    int bc = blockIdx.y;               // b*C + c
    int c  = bc % C;                   // uniform -> SALU
    int b  = bc / C;
    if (hw >= HW) return;
    int code = idx[b * HW + hw];
    q[(size_t)bc * HW + hw] = cb[(size_t)c * Kc + code];
}

__global__ void wnorm_kernel(const float* __restrict__ cb, float* __restrict__ wn,
                             int C, int Kc)
{
    int n = blockIdx.x * blockDim.x + threadIdx.x;
    if (n >= Kc) return;
    float s = 0.0f;
    for (int c = 0; c < C; ++c) { float v = cb[(size_t)c * Kc + n]; s += v * v; }
    wn[n] = s;
}

// ---------------------------------------------------------------------------
// Train-mode BatchNorm2d: biased batch stats + affine, optional fused ReLU.
// ---------------------------------------------------------------------------
__global__ void __launch_bounds__(256)
bn_stats_kernel(const float* __restrict__ x, float* __restrict__ mv,
                int C, int Bn, int HW)
{
    int c = blockIdx.x;
    float s = 0.0f, s2 = 0.0f;
    for (int b = 0; b < Bn; ++b) {
        const float* p = x + ((size_t)(b * C + c)) * HW;
        for (int i = threadIdx.x; i < HW; i += 256) {
            float v = p[i]; s += v; s2 += v * v;
        }
    }
    __shared__ float sh[256], sh2[256];
    sh[threadIdx.x] = s; sh2[threadIdx.x] = s2;
    __syncthreads();
    for (int o = 128; o > 0; o >>= 1) {
        if (threadIdx.x < o) {
            sh[threadIdx.x]  += sh[threadIdx.x + o];
            sh2[threadIdx.x] += sh2[threadIdx.x + o];
        }
        __syncthreads();
    }
    if (threadIdx.x == 0) {
        float n = (float)Bn * (float)HW;
        float m = sh[0] / n;
        mv[c] = m;
        mv[C + c] = sh2[0] / n - m * m;
    }
}

__global__ void bn_apply_kernel(const float* __restrict__ x, float* __restrict__ y,
                                const float* __restrict__ mv,
                                const float* __restrict__ g, const float* __restrict__ bta,
                                int C, int HW, int relu)
{
    int hw = blockIdx.x * blockDim.x + threadIdx.x;
    int bc = blockIdx.y;
    int c  = bc % C;                   // uniform -> SALU
    if (hw >= HW) return;
    size_t i = (size_t)bc * HW + hw;
    float m = mv[c], v = mv[C + c];
    float o = (x[i] - m) * rsqrtf(v + 1e-5f) * g[c] + bta[c];
    if (relu) o = o > 0.0f ? o : 0.0f;
    y[i] = o;
}

// align-corners bilinear resize + tanh; grid = (OH, Bn*C), loop over OW.
__global__ void resize_tanh_kernel(const float* __restrict__ x, float* __restrict__ y,
                                   int IH, int IW, int OH, int OW)
{
    int oy = blockIdx.x;
    int bc = blockIdx.y;
    const float* p = x + (size_t)bc * IH * IW;
    float*       q = y + (size_t)bc * OH * OW + (size_t)oy * OW;
    float fy = (float)oy * (float)(IH - 1) / (float)(OH - 1);
    int y0 = (int)floorf(fy);
    int y1 = y0 + 1 < IH ? y0 + 1 : IH - 1;
    float wy = fy - (float)y0;
    const float* r0 = p + (size_t)y0 * IW;
    const float* r1 = p + (size_t)y1 * IW;
    for (int ox = threadIdx.x; ox < OW; ox += blockDim.x) {
        float fx = (float)ox * (float)(IW - 1) / (float)(OW - 1);
        int x0 = (int)floorf(fx);
        int x1 = x0 + 1 < IW ? x0 + 1 : IW - 1;
        float wx = fx - (float)x0;
        float v = r0[x0] * (1.f - wy) * (1.f - wx) + r1[x0] * wy * (1.f - wx)
                + r0[x1] * (1.f - wy) * wx         + r1[x1] * wy * wx;
        q[ox] = tanhf(v);
    }
}

// ---------------------------------------------------------------------------
// Host-side launch helpers
// ---------------------------------------------------------------------------
static inline void launch_conv(hipStream_t st, const float* in, const float* w,
                               const float* bias, const float* res, float* out,
                               int Nb, int Cin, int Hin, int Win, int Cout,
                               int KH, int KW, int stride, int pad, int flags)
{
    int Hout, Wout;
    if (flags & CF_TRANS) { Hout = Hin * stride; Wout = Win * stride; }      // k=4,s=2,p=1
    else { Hout = (Hin + 2 * pad - KH) / stride + 1; Wout = (Win + 2 * pad - KW) / stride + 1; }
    int M = Nb * Hout * Wout;
    dim3 g((M + 15) / 16, (Cout + 127) / 128);
    conv2d_wmma_kernel<<<g, 128, 0, st>>>(in, w, bias, res, out,
        Nb, Cin, Hin, Win, Cout, Hout, Wout, KH, KW, stride, pad, flags);
}

static inline void launch_bn(hipStream_t st, const float* x, float* y,
                             const float* g, const float* b, float* mv,
                             int C, int Bn, int HW, int relu)
{
    bn_stats_kernel<<<C, 256, 0, st>>>(x, mv, C, Bn, HW);
    bn_apply_kernel<<<dim3((HW + 255) / 256, Bn * C), 256, 0, st>>>(
        x, y, mv, g, b, C, HW, relu);
}

extern "C" void kernel_launch(void* const* d_in, const int* in_sizes, int n_in,
                              void* d_out, int out_size, void* d_ws, size_t ws_size,
                              hipStream_t stream)
{
    (void)in_sizes; (void)n_in; (void)out_size; (void)ws_size;
    const float* x = (const float*)d_in[0];
    #define P(i) ((const float*)d_in[i])
    float* out = (float*)d_out;
    float* ws  = (float*)d_ws;

    const int B = 4, HID = 128, D = 256, K = 512;
    const int HW112 = 112 * 112, HW56 = 56 * 56, HW224 = 224 * 224;
    const size_t SZ56_H = (size_t)B * HID * HW56;     // 1,605,632

    // ---- output layout (concatenated in return order) ----
    float* out_full = out;                                   // (4,3,224,224)
    float* out_half = out + 602112;                          // (4,1,112,112)
    float* ze_out   = out + 652288;                          // (4,256,56,56)
    float* emb_out  = out + 3863552;                         // (4,256,56,56)
    float* zea_out  = out + 7074816;                         // (4,128,56,56)
    float* emba_out = out + 8680448;                         // (4,128,56,56)
    int*   arg_out  = (int*)(out + 10286080);                // (4,112,112)

    // ---- workspace layout ----
    float* s0 = ws;                         // 25,690,112 : t1 / dza / xb
    float* s1 = s0 + 25690112;              // 12,845,056 : t8 / t12 / zq
    float* s2 = s1 + 12845056;              // 12,845,056 : decoder temps
    float* s3 = s2 + 12845056;              //  2,408,448 : output-head convT
    float* s4 = s3 + 2408448;               //  6,422,528 : encoder temps
    float* mv  = s4 + 6422528;              // 512 (mean|var)
    float* wn0 = mv + 512;                  // 512
    float* wn1 = wn0 + 512;                 // 512
    int*   idxb = (int*)(wn1 + 512);        // 12,544 ints

    // ================= encoder_a =================
    float* t1 = s0;                                                   // (4,128,112,112)
    launch_conv(stream, x, P(1), P(2), nullptr, t1, B, 3, 224, 224, HID, 4, 4, 2, 1, 0);
    launch_bn(stream, t1, t1, P(3), P(4), mv, HID, B, HW112, 1);
    float* t3 = s4;                                                   // (4,128,56,56)
    launch_conv(stream, t1, P(5), P(6), nullptr, t3, B, HID, 112, 112, HID, 4, 4, 2, 1, 0);
    launch_bn(stream, t3, t3, P(7), P(8), mv, HID, B, HW56, 1);
    float* t5 = s4 + SZ56_H;
    launch_conv(stream, t3, P(9), P(10), nullptr, t5, B, HID, 56, 56, HID, 3, 3, 1, 1, CF_RELU_IN);
    launch_bn(stream, t5, t5, P(11), P(12), mv, HID, B, HW56, 1);
    float* t7 = s4 + 2 * SZ56_H;
    launch_conv(stream, t5, P(13), P(14), t3, t7, B, HID, 56, 56, HID, 1, 1, 1, 0, CF_ADD_RES);
    launch_bn(stream, t7, zea_out, P(15), P(16), mv, HID, B, HW56, 0);

    // ================= VQ stage 0 (emb0) =================
    wnorm_kernel<<<2, 256, 0, stream>>>(P(17), wn0, HID, K);
    vq_argmin_kernel<<<(B * HW56 + 15) / 16, 32, 0, stream>>>(zea_out, P(17), wn0, B, HID, HW56, K, idxb);
    vq_gather_kernel<<<dim3((HW56 + 255) / 256, B * HID), 256, 0, stream>>>(P(17), idxb, emba_out, HID, HW56, K);
    // zq_a (straight-through forward value) == emba_out

    // ================= encoder_b =================
    float* t8 = s1;                                                   // (4,256,56,56)
    launch_conv(stream, emba_out, P(18), P(19), nullptr, t8, B, HID, 56, 56, D, 3, 3, 1, 1, CF_RELU_IN);
    launch_bn(stream, t8, t8, P(20), P(21), mv, D, B, HW56, 1);
    float* t8b = s1 + 3211264;
    launch_conv(stream, t8, P(22), P(23), nullptr, t8b, B, D, 56, 56, D, 1, 1, 1, 0, 0);
    launch_bn(stream, t8b, ze_out, P(24), P(25), mv, D, B, HW56, 0);

    // ================= emb = VQ(ze, emb1) =================
    wnorm_kernel<<<2, 256, 0, stream>>>(P(40), wn1, D, K);
    vq_argmin_kernel<<<(B * HW56 + 15) / 16, 32, 0, stream>>>(ze_out, P(40), wn1, B, D, HW56, K, idxb);
    vq_gather_kernel<<<dim3((HW56 + 255) / 256, B * D), 256, 0, stream>>>(P(40), idxb, emb_out, D, HW56, K);

    // ================= decoder_a =================
    float* t9  = s2;                                                  // (4,128,56,56)
    launch_conv(stream, ze_out, P(26), P(27), nullptr, t9, B, D, 56, 56, HID, 3, 3, 1, 1, CF_RELU_IN);
    float* t10 = s2 + SZ56_H;
    launch_conv(stream, t9, P(28), P(29), nullptr, t10, B, HID, 56, 56, HID, 1, 1, 1, 0, CF_RELU_IN);
    launch_bn(stream, t10, t10, P(30), P(31), mv, HID, B, HW56, 0);
    float* t11 = s2 + 2 * SZ56_H;                                     // (4,128,112,112)
    launch_conv(stream, t10, P(32), P(33), nullptr, t11, B, HID, 56, 56, HID, 4, 4, 2, 1, CF_TRANS);
    launch_bn(stream, t11, t11, P(34), P(35), mv, HID, B, HW112, 1);
    float* t12 = s1;                                                  // (4,256,112,112)
    launch_conv(stream, t11, P(36), P(37), nullptr, t12, B, HID, 112, 112, D, 3, 3, 1, 1, CF_RELU_IN);
    float* dza = s0;                                                  // (4,256,112,112)
    launch_conv(stream, t12, P(38), P(39), nullptr, dza, B, D, 112, 112, D, 1, 1, 1, 0, CF_RELU_IN);

    // ================= VQ stage 1 (emb1) -> argmin output =================
    vq_argmin_kernel<<<(B * HW112 + 15) / 16, 32, 0, stream>>>(dza, P(40), wn1, B, D, HW112, K, arg_out);
    float* zq = s1;                                                   // (4,256,112,112)
    vq_gather_kernel<<<dim3((HW112 + 255) / 256, B * D), 256, 0, stream>>>(P(40), arg_out, zq, D, HW112, K);

    // ================= decode: xa0 branch =================
    float* xh  = s2;                                                  // (4,128,112,112)
    launch_conv(stream, zq, P(41), P(42), nullptr, xh, B, D, 112, 112, HID, 3, 3, 1, 1, CF_RELU_IN);
    float* xa0 = s2 + 6422528;                                        // (4,128,112,112)
    launch_conv(stream, xh, P(43), P(44), nullptr, xa0, B, HID, 112, 112, HID, 1, 1, 1, 0, CF_RELU_IN);
    launch_bn(stream, xa0, xa0, P(45), P(46), mv, HID, B, HW112, 1);

    // ================= decode: xb branch =================
    float* xb = s0;                                                   // (4,128,224,224)
    launch_conv(stream, zq, P(47), P(48), nullptr, xb, B, D, 112, 112, HID, 4, 4, 2, 1, CF_TRANS);
    launch_bn(stream, xb, xb, P(49), P(50), mv, HID, B, HW224, 1);

    // out_full: ConvT(xb, of) -> (4,3,448,448) -> bilinear(224) -> tanh
    launch_conv(stream, xb, P(51), P(52), nullptr, s3, B, HID, 224, 224, 3, 4, 4, 2, 1, CF_TRANS);
    resize_tanh_kernel<<<dim3(224, B * 3), 256, 0, stream>>>(s3, out_full, 448, 448, 224, 224);

    // out_half: ConvT(xa0, oh) -> (4,1,224,224) -> bilinear(112) -> tanh
    launch_conv(stream, xa0, P(53), P(54), nullptr, s3, B, HID, 112, 112, 1, 4, 4, 2, 1, CF_TRANS);
    resize_tanh_kernel<<<dim3(112, B * 1), 256, 0, stream>>>(s3, out_half, 224, 224, 112, 112);
    #undef P
}
